// XYLoss_64012192580021
// MI455X (gfx1250) — matrix operand order repaired and checked by reading, not compiled, not run on hardware
//
#include <hip/hip_runtime.h>
#include <math.h>

// ---------------- constants from the reference ----------------
#define NUM_CLASS 100000
#define EMB       512
#define NROWS     1024
#define S_SCALE   64.0f
#define T_PLUS1   1.2f           // T + 1
#define NUM_ALL   2900.0f
#define EMA       0.01f
#define CLEAR_RIO 0.7f           // 1 - 0.3
// m2 = 0.5, m3 = 0.2 margin constants (precomputed)
#define COS_M2   0.87758256189037276f   // cos(0.5)
#define SIN_M2   0.47942553860420301f   // sin(0.5)
#define THETA   (-0.87758256189037276f) // cos(pi-0.5)
#define SINMM    0.23971276930210150f   // sin(pi-0.5)*0.5
#define COS_M3   0.98006657784124163f   // cos(0.2)
#define SIN_M3   0.19866933079506122f   // sin(0.2)
#define SINMM_  (-0.03973386615901224f) // sin(-0.2)*0.2

typedef __attribute__((ext_vector_type(16))) _Float16 v16h;
typedef __attribute__((ext_vector_type(8)))  _Float16 v8h;
typedef __attribute__((ext_vector_type(8)))  float    v8f;
typedef __attribute__((ext_vector_type(4)))  float    v4f;   // true clang vector
                                                             // (NT builtins OK)

// ---------------------------------------------------------------------------
// Kernel 0: per-row stats for logits. One block per row (256 thr, 2 elem/thr).
//  - row L2 norm of logits; write normalized logits as f16 (WMMA A operand)
//  - gather weight[label], compute tgt = <nl, nw>, then ftl / ftl_
//  - zero the global hard counter
// ---------------------------------------------------------------------------
__global__ __launch_bounds__(256) void row_stats_kernel(
    const float* __restrict__ logits, const int* __restrict__ labels,
    const float* __restrict__ weight, float* __restrict__ ftl,
    float* __restrict__ ftl2, _Float16* __restrict__ nlH,
    unsigned int* __restrict__ hard_cnt) {
  const int row = blockIdx.x;
  const int tid = threadIdx.x;
  if (row == 0 && tid == 0) *hard_cnt = 0u;

  const float* xr = logits + (size_t)row * EMB;
  const int lab = labels[row];
  const float* wr = weight + (size_t)lab * EMB;

  float x0 = xr[tid], x1 = xr[tid + 256];
  float w0 = wr[tid], w1 = wr[tid + 256];

  __shared__ float sX[256], sW[256], sD[256];
  sX[tid] = x0 * x0 + x1 * x1;
  sW[tid] = w0 * w0 + w1 * w1;
  sD[tid] = x0 * w0 + x1 * w1;
  __syncthreads();
  for (int s = 128; s > 0; s >>= 1) {
    if (tid < s) { sX[tid] += sX[tid+s]; sW[tid] += sW[tid+s]; sD[tid] += sD[tid+s]; }
    __syncthreads();
  }
  const float invx = rsqrtf(sX[0] + 1e-12f);
  const float invw = rsqrtf(sW[0] + 1e-12f);

  nlH[(size_t)row * EMB + tid]       = (_Float16)(x0 * invx);
  nlH[(size_t)row * EMB + tid + 256] = (_Float16)(x1 * invx);

  if (tid == 0) {
    float tgt = fminf(fmaxf(sD[0] * invx * invw, -1.0f), 1.0f);
    float st  = sqrtf(fmaxf(1.0f - tgt * tgt, 0.0f));
    float ctm  = tgt * COS_M2 - st * SIN_M2;
    float ctm_ = tgt * COS_M3 - st * SIN_M3;
    ftl [row] = (tgt >  THETA)  ? ctm  : (tgt - SINMM);
    ftl2[row] = (tgt <= COS_M3) ? ctm_ : (tgt + SINMM_);
  }
}

// ---------------------------------------------------------------------------
// Kernel 1: normalize every weight row -> f16 (WMMA B operand, row-major in K)
// One block per row, 256 threads, 2 elements each.
// ---------------------------------------------------------------------------
__global__ __launch_bounds__(256) void wnorm_kernel(
    const float* __restrict__ weight, _Float16* __restrict__ nwH) {
  const int row = blockIdx.x;
  const int tid = threadIdx.x;
  const float* wr = weight + (size_t)row * EMB;
  float a = wr[tid], b = wr[tid + 256];

  __shared__ float sS[256];
  sS[tid] = a * a + b * b;
  __syncthreads();
  for (int s = 128; s > 0; s >>= 1) {
    if (tid < s) sS[tid] += sS[tid + s];
    __syncthreads();
  }
  const float inv = rsqrtf(sS[0] + 1e-12f);
  nwH[(size_t)row * EMB + tid]       = (_Float16)(a * inv);
  nwH[(size_t)row * EMB + tid + 256] = (_Float16)(b * inv);
}

// ---------------------------------------------------------------------------
// Kernel 2: WMMA GEMM. cos = clip(A @ B^T), store to out, count cos > ftl[row].
// Block = 256 threads = 8 waves; each wave computes a 16(M) x 32(N) tile with
// two v_wmma_f32_16x16x32_f16 per K-step; K = 512 -> 16 steps.
// Grid = (1024/128, 100000/32) = (8, 3125), exact cover.
// All 8 x-blocks of a given column tile share the same 32 B-columns -> B
// (102.4 MB f16) is L2-resident; cos stores are non-temporal so the 410 MB
// output stream does not evict B from the 192 MB L2.
// ---------------------------------------------------------------------------
__global__ __launch_bounds__(256) void gemm_kernel(
    const _Float16* __restrict__ A,      // [1024][512] normalized logits
    const _Float16* __restrict__ B,      // [100000][512] normalized weight
    const float* __restrict__ ftl,
    float* __restrict__ out,             // [1024][100000] clamped cos
    unsigned int* __restrict__ hard_cnt) {
  const int lane = threadIdx.x & 31;
  const int wave = threadIdx.x >> 5;
  const int hw   = lane >> 4;            // half-wave: 0 or 1
  const int l16  = lane & 15;

  const int rowTile = blockIdx.x * 128 + wave * 16;
  const int colTile = blockIdx.y * 32;

  // A fragment layout: lane holds row M = l16;
  // halves 0..15 = K hw*16 .. hw*16+15 (matches B).
  const _Float16* aPtr  = A + (size_t)(rowTile + l16) * EMB + hw * 16;
  // B fragment layout (rows striped across lanes): lane holds col N = l16;
  // halves 0..15 = K hw*16 .. hw*16+15 -> one contiguous 32B load per lane.
  const _Float16* b0Ptr = B + (size_t)(colTile + l16)      * EMB + hw * 16;
  const _Float16* b1Ptr = B + (size_t)(colTile + 16 + l16) * EMB + hw * 16;

  v8f c0 = {}; v8f c1 = {};
  for (int k = 0; k < EMB; k += 32) {
    v8h alo = *(const v8h*)(aPtr + k);
    v8h ahi = *(const v8h*)(aPtr + k + 8);
    v16h afrag = __builtin_shufflevector(alo, ahi,
        0,1,2,3,4,5,6,7,8,9,10,11,12,13,14,15);
    v16h b0 = *(const v16h*)(b0Ptr + k);
    v16h b1 = *(const v16h*)(b1Ptr + k);
    c0 = __builtin_amdgcn_wmma_f32_16x16x32_f16(false, afrag, false, b0,
                                                (short)0, c0, false, false);
    c1 = __builtin_amdgcn_wmma_f32_16x16x32_f16(false, afrag, false, b1,
                                                (short)0, c1, false, false);
  }

  // C/D layout: VGPR r -> M = r + 8*hw, N = l16.
  unsigned int cnt = 0;
  const int col0 = colTile + l16;
  const int col1 = col0 + 16;
  #pragma unroll
  for (int r = 0; r < 8; ++r) {
    const int row = rowTile + r + hw * 8;
    const float f = ftl[row];
    float v0 = fminf(fmaxf(c0[r], -1.0f), 1.0f);
    float v1 = fminf(fmaxf(c1[r], -1.0f), 1.0f);
    cnt += (v0 > f) ? 1u : 0u;
    cnt += (v1 > f) ? 1u : 0u;
    __builtin_nontemporal_store(v0, &out[(size_t)row * NUM_CLASS + col0]);
    __builtin_nontemporal_store(v1, &out[(size_t)row * NUM_CLASS + col1]);
  }

  __shared__ unsigned int sc;
  if (threadIdx.x == 0) sc = 0u;
  __syncthreads();
  atomicAdd(&sc, cnt);
  __syncthreads();
  if (threadIdx.x == 0) atomicAdd(hard_cnt, sc);
}

// ---------------------------------------------------------------------------
// Kernel 3: memory-bound finalize. v4f per thread; 100000 blocks x 256 thr
// covers 102.4M elements exactly (NUM_CLASS % 4 == 0 -> no row crossing).
// Pure streaming pass (zero reuse) -> non-temporal load + store.
// ---------------------------------------------------------------------------
__global__ __launch_bounds__(256) void finalize_kernel(
    float* __restrict__ out, const int* __restrict__ labels,
    const float* __restrict__ ftl, const float* __restrict__ ftl2,
    const unsigned int* __restrict__ hard_cnt) {
  const float hard = (float)(*hard_cnt);
  const float m_smooth = ((NUM_ALL - hard) / NUM_ALL) * EMA;   // prev = 0
  const float tns = 1.0f - m_smooth / CLEAR_RIO;
  const float noise_scale = tns * tns;                         // GAMMA = 2

  const size_t i4   = (size_t)blockIdx.x * 256 + threadIdx.x;
  const size_t base = i4 * 4;
  const int row = (int)(base / NUM_CLASS);
  const int col = (int)(base % NUM_CLASS);
  const float f  = ftl[row];
  const float f_ = ftl2[row];
  const int   lab = labels[row];

  v4f v = __builtin_nontemporal_load((const v4f*)(out + base));
  #pragma unroll
  for (int j = 0; j < 4; ++j) {
    float cv = v[j];
    float o  = (cv > f) ? cv * T_PLUS1 : cv;                 // hard mask
    if (cv > f_) o = fmaxf(cv * noise_scale, 1e-30f);        // noise mask wins
    if (col + j == lab) o = f;                               // label override
    v[j] = o * S_SCALE;
  }
  __builtin_nontemporal_store(v, (v4f*)(out + base));
}

// ---------------------------------------------------------------------------
extern "C" void kernel_launch(void* const* d_in, const int* in_sizes, int n_in,
                              void* d_out, int out_size, void* d_ws, size_t ws_size,
                              hipStream_t stream) {
  (void)in_sizes; (void)n_in; (void)out_size; (void)ws_size;
  const float* logits = (const float*)d_in[0];
  const int*   labels = (const int*)d_in[1];
  const float* weight = (const float*)d_in[2];
  float* out = (float*)d_out;

  // workspace layout (all 256B aligned):
  //  [0]        hard counter (u32)
  //  [256]      ftl  [1024] f32
  //  [4352]     ftl_ [1024] f32
  //  [8448]     normalized logits f16  [1024*512]   (1 MB)
  //  [1056]KB+  normalized weight f16 [100000*512]  (102.4 MB)
  char* ws = (char*)d_ws;
  unsigned int* hard = (unsigned int*)ws;
  float* ftl  = (float*)(ws + 256);
  float* ftl2 = (float*)(ws + 256 + 4096);
  _Float16* nlH = (_Float16*)(ws + 8448);
  _Float16* nwH = (_Float16*)(ws + 8448 + (size_t)NROWS * EMB * sizeof(_Float16));

  row_stats_kernel<<<NROWS, 256, 0, stream>>>(logits, labels, weight,
                                              ftl, ftl2, nlH, hard);
  wnorm_kernel<<<NUM_CLASS, 256, 0, stream>>>(weight, nwH);

  dim3 g2(NROWS / 128, NUM_CLASS / 32);         // (8, 3125)
  gemm_kernel<<<g2, 256, 0, stream>>>(nlH, nwH, ftl, out, hard);

  finalize_kernel<<<NUM_CLASS, 256, 0, stream>>>(out, labels, ftl, ftl2, hard);
}
